// VSSM_13752485282426
// MI455X (gfx1250) — compile-verified, matrix operand order checked by reading, not run on hardware
//
#include <hip/hip_runtime.h>
#include <hip/hip_bf16.h>

// ---------------- problem constants (from reference setup_inputs) ------------
#define Bb   2
#define Dd   192
#define Hh   64
#define Ww   64
#define Ll   (Hh * Ww)          // 4096
#define Kk   8
#define BK   (Bb * Kk)          // 16
#define Nn   16
#define N2   (2 * Nn)           // combined B|C rows
#define Rr   6
#define Cc   (Rr + 2 * Nn)      // 38
#define CPAD 48                 // C padded to 3 x 16 WMMA M-tiles
#define LT   64                 // L tile per proj block (4 N-tiles, 4 waves)
#define NC   16                 // number of scan chunks
#define CL   (Ll / NC)          // 256 steps per chunk

typedef __attribute__((ext_vector_type(16))) __bf16 v16bf;
typedef __attribute__((ext_vector_type(8)))  float  v8f;
typedef __attribute__((ext_vector_type(4)))  unsigned int u32x4;
typedef __attribute__((ext_vector_type(8)))  unsigned int u32x8v;
typedef __attribute__((ext_vector_type(8)))  int    i32x8;
typedef __attribute__((ext_vector_type(4)))  int    i32x4;

// ------------- cross-scan permutation: source index into H*W per (k,l) ------
__device__ __forceinline__ int sidx_k(int k, int l) {
    int kk = k;
    if (k == 2) { l = Ll - 1 - l; kk = 0; }
    else if (k == 3) { l = Ll - 1 - l; kk = 1; }
    else if (k == 6) { l = Ll - 1 - l; kk = 4; }
    else if (k == 7) { l = Ll - 1 - l; kk = 5; }
    if (kk == 0) return l;
    int h = l & (Hh - 1), w = l >> 6;           // l = w*H + h
    if (kk == 1) return h * Ww + w;
    if (kk == 4) return h * Ww + ((h + w) & (Ww - 1));
    return h * Ww + ((w - h) & (Ww - 1));       // kk == 5
}

// ------------- cross-merge inverse permutation per (k, output l) -------------
__device__ __forceinline__ int pinv_k(int k, int l) {
    int h = l >> 6, w = l & (Ww - 1);           // l = h*W + w
    int base = k, rev = 0;
    if (k == 2) { base = 0; rev = 1; }
    else if (k == 3) { base = 1; rev = 1; }
    else if (k == 6) { base = 4; rev = 1; }
    else if (k == 7) { base = 5; rev = 1; }
    int p;
    if (base == 0)      p = l;
    else if (base == 1) p = w * Hh + h;
    else if (base == 4) p = ((w - h) & (Ww - 1)) * Hh + h;
    else                p = ((w + h) & (Ww - 1)) * Hh + h;
    return rev ? (Ll - 1 - p) : p;
}

// ------------- TDM: 1-D contiguous fp32 tile, global -> LDS (D# by hand) ----
__device__ __forceinline__ void tdm_load_1d_f32(const float* gptr, float* ldsptr,
                                                unsigned nelem) {
    unsigned long long ga = (unsigned long long)(size_t)gptr;
    unsigned lds = (unsigned)(size_t)ldsptr;   // flat LDS addr[31:0] == LDS byte offset
    u32x4 g0;
    g0.x = 1u;                                  // count=1, is_restore=0, gather off
    g0.y = lds;                                 // lds_addr [63:32]
    g0.z = (unsigned)(ga & 0xFFFFFFFFull);      // global_addr lo
    g0.w = (unsigned)((ga >> 32) & 0x1FFFFFFull) | (2u << 30); // addr hi | type=2
    unsigned long long td0 = nelem, st0 = nelem;
    i32x8 g1;
    g1[0] = (int)(2u << 16);                    // workgroup_mask=0, data_size=2 (4B)
    g1[1] = (int)((td0 & 0xFFFFull) << 16);     // tensor_dim0 [79:48] low half
    g1[2] = (int)(((td0 >> 16) & 0xFFFFull) | (1u << 16)); // dim0 hi | tensor_dim1=1
    g1[3] = (int)((unsigned)(nelem & 0xFFFFu) << 16);       // tile_dim0
    g1[4] = 0;                                  // tile_dim1=0, tile_dim2=0 (unused)
    g1[5] = (int)(st0 & 0xFFFFFFFFull);         // tensor_dim0_stride lo
    g1[6] = (int)((st0 >> 32) & 0xFFFFull);     // stride0 hi | stride1=0
    g1[7] = 0;
    i32x4 g2 = {0, 0, 0, 0}, g3 = {0, 0, 0, 0};
    i32x8 g4 = {0, 0, 0, 0, 0, 0, 0, 0};
    __builtin_amdgcn_tensor_load_to_lds(g0, g1, g2, g3, g4, 0);
}

// pack two f32 into one dword of two bf16 (lo = a, hi = b)
__device__ __forceinline__ unsigned pack_bf16(float a, float b) {
    unsigned short ua = __builtin_bit_cast(unsigned short, (__bf16)a);
    unsigned short ub = __builtin_bit_cast(unsigned short, (__bf16)b);
    return (unsigned)ua | ((unsigned)ub << 16);
}

// Build one 16-bf16 WMMA fragment from a 16-dword pair-packed LDS block:
// lane needs dwords {half*4 .. half*4+3} and {8+half*4 .. 8+half*4+3}
// -> two ds_load_b128 per operand.
__device__ __forceinline__ v16bf frag_from(const unsigned int* blk, int half) {
    const uint4* p = (const uint4*)blk;
    uint4 lo = p[half];
    uint4 hi = p[2 + half];
    u32x8v u = {lo.x, lo.y, lo.z, lo.w, hi.x, hi.y, hi.z, hi.w};
    return __builtin_bit_cast(v16bf, u);
}

// ---------------------------------------------------------------------------
// Kernel 1: fused cross-scan gather + projection GEMMs (WMMA bf16 -> f32).
//   Per block: one (b,k), one LT=64 column tile of L. 128 threads (4 waves).
//   Stage 1: x_dbl = Wk(38x192) @ xs(192x64)  -> dts_raw | B | C
//   Stage 2: dts   = dtw(192x6) @ dts_raw(6x64), +bias, softplus -> delta
// LDS (dwords): lwp 4608 (overlaid by lxd 3072 after stage 1) | lxp 6144 |
//               ldtbp 1024 | ldwp 3072   => 59.4 KB static
// ---------------------------------------------------------------------------
__global__ __launch_bounds__(128) void vssm_proj_kernel(
    const float* __restrict__ x,    // (B,D,H,W)
    const float* __restrict__ xw,   // (K,38,D)
    const float* __restrict__ dtw,  // (K,D,R)
    const float* __restrict__ dtb,  // (K,D)
    float* __restrict__ xs_t,       // (B,K,L,D)
    float* __restrict__ delta_t,    // (B,K,L,D)
    float* __restrict__ bc_t)       // (B,K,L,2N)  rows 0..15 = B, 16..31 = C
{
    __shared__ __align__(16) unsigned int smem[14848];
    unsigned int* lwp   = smem;                         // 48 rows x 96 dwords
    float*        lxd   = (float*)smem;                 // overlay: 48 x 64 f32
    unsigned int* lxp   = smem + 4608;                  // 64 cols x 96 dwords
    unsigned int* ldtbp = smem + 4608 + 6144;           // 64 cols x 16 dwords
    unsigned int* ldwp  = smem + 4608 + 6144 + 1024;    // 192 rows x 16 dwords

    const int blk = blockIdx.x;
    const int ltile = blk % (Ll / LT);
    const int bk = blk / (Ll / LT);
    const int b = bk / Kk, k = bk % Kk;
    const int tid = threadIdx.x;
    const int l0 = ltile * LT;

    // --- stage A-weights (pair-packed bf16), zero-padded rows 38..47 ---
    for (int i = tid; i < CPAD * (Dd / 2); i += 128) {
        int r = i / (Dd / 2), t = i % (Dd / 2);
        float a = 0.f, bb = 0.f;
        if (r < Cc) {
            const float* row = xw + (size_t)(k * Cc + r) * Dd;
            a = row[2 * t]; bb = row[2 * t + 1];
        }
        lwp[i] = pack_bf16(a, bb);
    }
    // --- stage dt-weights (192 x K32, pair-packed, K>=6 zero) ---
    for (int i = tid; i < Dd * 16; i += 128) {
        int d = i / 16, t = i % 16;
        float a = (2 * t     < Rr) ? dtw[(k * Dd + d) * Rr + 2 * t]     : 0.f;
        float bb = (2 * t + 1 < Rr) ? dtw[(k * Dd + d) * Rr + 2 * t + 1] : 0.f;
        ldwp[i] = pack_bf16(a, bb);
    }
    // --- gather cross-scan tile: column-major packed bf16 + fp32 xs_t ---
    const float* xb = x + (size_t)b * Dd * Ll;
    for (int i = tid; i < LT * (Dd / 2); i += 128) {
        int li = i % LT, t = i / LT;                 // li fastest: coalesced in l
        int src = sidx_k(k, l0 + li);
        float a = xb[(size_t)(2 * t) * Ll + src];
        float bb = xb[(size_t)(2 * t + 1) * Ll + src];
        lxp[li * 96 + t] = pack_bf16(a, bb);
        float2 v2; v2.x = a; v2.y = bb;
        *(float2*)&xs_t[((size_t)bk * Ll + l0 + li) * Dd + 2 * t] = v2;
    }
    __syncthreads();

    const int lane = tid & 31;
    const int half = lane >> 4, l16 = lane & 15;
    const int ncol = (tid >> 5) * 16;            // each wave owns one 16-col N-tile
    const unsigned int* bblk = lxp + (ncol + l16) * 96;

    // ---------------- stage 1: x_dbl = Wk @ xs_tile (3 M-tiles kept live) ----
    v8f acc0 = {}, acc1 = {}, acc2 = {};
    #pragma unroll
    for (int ks = 0; ks < 6; ++ks) {             // K = 192 = 6 * 32
        v16bf bfr = frag_from(bblk + ks * 16, half);
        v16bf a0 = frag_from(lwp + (0 * 16 + l16) * 96 + ks * 16, half);
        acc0 = __builtin_amdgcn_wmma_f32_16x16x32_bf16(false, a0, false, bfr,
                                                       (short)0, acc0, false, false);
        v16bf a1 = frag_from(lwp + (1 * 16 + l16) * 96 + ks * 16, half);
        acc1 = __builtin_amdgcn_wmma_f32_16x16x32_bf16(false, a1, false, bfr,
                                                       (short)0, acc1, false, false);
        v16bf a2 = frag_from(lwp + (2 * 16 + l16) * 96 + ks * 16, half);
        acc2 = __builtin_amdgcn_wmma_f32_16x16x32_bf16(false, a2, false, bfr,
                                                       (short)0, acc2, false, false);
    }
    __syncthreads();                             // done reading lwp -> reuse as lxd

    // park all x_dbl rows in f32 LDS tile (row = C index, col = l within tile)
    #pragma unroll
    for (int i = 0; i < 8; ++i) {
        int r0 = i + 8 * half;
        lxd[(r0     ) * LT + ncol + l16] = acc0[i];
        lxd[(r0 + 16) * LT + ncol + l16] = acc1[i];
        lxd[(r0 + 32) * LT + ncol + l16] = acc2[i];
    }
    __syncthreads();

    // cooperative coalesced write of B|C rows 6..37 -> bc_t (b,k,l,32)
    for (int i = tid; i < N2 * LT; i += 128) {
        int n2 = i % N2, li = i / N2;
        bc_t[((size_t)bk * Ll + l0 + li) * N2 + n2] = lxd[(Rr + n2) * LT + li];
    }
    // repack dts_raw rows 0..5 (K-padded to 32) into pair-packed bf16
    for (int i = tid; i < LT * 16; i += 128) {
        int li = i / 16, t = i % 16;
        float a = (2 * t     < Rr) ? lxd[(2 * t    ) * LT + li] : 0.f;
        float bb = (2 * t + 1 < Rr) ? lxd[(2 * t + 1) * LT + li] : 0.f;
        ldtbp[li * 16 + t] = pack_bf16(a, bb);
    }
    __syncthreads();

    // ------------- stage 2: dts = dtw @ dts_raw, bias + softplus -------------
    const unsigned int* b2blk = ldtbp + (ncol + l16) * 16;
    v16bf bfr2 = frag_from(b2blk, half);
    for (int mt = 0; mt < 12; ++mt) {            // M = 192 = 12 * 16
        v16bf a = frag_from(ldwp + (mt * 16 + l16) * 16, half);
        v8f acc = {};
        acc = __builtin_amdgcn_wmma_f32_16x16x32_bf16(false, a, false, bfr2,
                                                      (short)0, acc, false, false);
        float ov[8];
        #pragma unroll
        for (int i = 0; i < 8; ++i) {
            int d = mt * 16 + i + 8 * half;
            float v = acc[i] + dtb[k * Dd + d];
            ov[i] = (v > 20.f) ? v : log1pf(__expf(v));   // softplus
        }
        // lane writes 8 consecutive d -> two aligned b128 stores
        float* dst = &delta_t[((size_t)bk * Ll + l0 + ncol + l16) * Dd
                              + mt * 16 + 8 * half];
        float4 w0; w0.x = ov[0]; w0.y = ov[1]; w0.z = ov[2]; w0.w = ov[3];
        float4 w1; w1.x = ov[4]; w1.y = ov[5]; w1.z = ov[6]; w1.w = ov[7];
        ((float4*)dst)[0] = w0;
        ((float4*)dst)[1] = w1;
    }
}

// ---------------------------------------------------------------------------
// Kernel 2 (scan phase 1): per (b,k,chunk) compute the chunk's linear transfer
//   hA = prod_t dA_t,  hB = state reached from h_in = 0. 192 threads = one d.
//   Combined B|C chunk (CL x 32 f32, contiguous) streamed into LDS by the TDM.
// ---------------------------------------------------------------------------
__global__ __launch_bounds__(192) void vssm_scan_phase1(
    const float* __restrict__ xs_t, const float* __restrict__ delta_t,
    const float* __restrict__ bc_t, const float* __restrict__ A_logs,
    float* __restrict__ chunkA, float* __restrict__ chunkB)
{
    __shared__ float lbc[CL * N2];               // 32 KB chunk of B|C
    const int blk = blockIdx.x;
    const int c = blk % NC, bk = blk / NC;
    const int d = threadIdx.x;

    if (threadIdx.x < 32) {                      // wave 0 drives the tensor DMA
        tdm_load_1d_f32(bc_t + ((size_t)bk * Ll + c * CL) * N2, lbc, CL * N2);
        __builtin_amdgcn_s_wait_tensorcnt(0);
    }
    __syncthreads();

    const int kd = (bk % Kk) * Dd + d;
    float Av[Nn];
    #pragma unroll
    for (int n = 0; n < Nn; ++n) Av[n] = -__expf(A_logs[kd * Nn + n]);

    float hA[Nn], hB[Nn];
    #pragma unroll
    for (int n = 0; n < Nn; ++n) { hA[n] = 1.f; hB[n] = 0.f; }

    size_t base = ((size_t)bk * Ll + c * CL) * Dd + d;
    for (int t = 0; t < CL; ++t) {
        float dlt = delta_t[base + (size_t)t * Dd];   // coalesced across d
        float u   = xs_t[base + (size_t)t * Dd];
        float du  = dlt * u;
        const float* bt = lbc + t * N2;
        #pragma unroll
        for (int n = 0; n < Nn; ++n) {
            float dA = __expf(dlt * Av[n]);
            hA[n] *= dA;
            hB[n] = hB[n] * dA + du * bt[n];
        }
    }
    size_t off = (((size_t)bk * NC + c) * Dd + d) * Nn;
    #pragma unroll
    for (int n = 0; n < Nn; ++n) { chunkA[off + n] = hA[n]; chunkB[off + n] = hB[n]; }
}

// ---------------------------------------------------------------------------
// Kernel 3 (scan phase 2): tiny sequential combine across the 16 chunks;
//   one thread per (b,k,d,n) channel. Emits h_in for every chunk.
// ---------------------------------------------------------------------------
__global__ void vssm_scan_phase2(const float* __restrict__ chunkA,
                                 const float* __restrict__ chunkB,
                                 float* __restrict__ hin)
{
    int idx = blockIdx.x * blockDim.x + threadIdx.x;
    if (idx >= BK * Dd * Nn) return;
    int bk = idx / (Dd * Nn);
    int dn = idx % (Dd * Nn);
    float h = 0.f;
    for (int c = 0; c < NC; ++c) {
        size_t off = ((size_t)bk * NC + c) * (Dd * Nn) + dn;
        hin[off] = h;
        h = chunkA[off] * h + chunkB[off];
    }
}

// ---------------------------------------------------------------------------
// Kernel 4 (scan phase 3): replay each chunk with its true h_in, produce
//   y = C.h + u*Ds, stored transposed (b,k,l,d). B|C chunk via one TDM load.
// ---------------------------------------------------------------------------
__global__ __launch_bounds__(192) void vssm_scan_phase3(
    const float* __restrict__ xs_t, const float* __restrict__ delta_t,
    const float* __restrict__ bc_t, const float* __restrict__ A_logs,
    const float* __restrict__ Ds, const float* __restrict__ hin,
    float* __restrict__ ys_t)
{
    __shared__ float lbc[CL * N2];
    const int blk = blockIdx.x;
    const int c = blk % NC, bk = blk / NC;
    const int d = threadIdx.x;

    if (threadIdx.x < 32) {
        tdm_load_1d_f32(bc_t + ((size_t)bk * Ll + c * CL) * N2, lbc, CL * N2);
        __builtin_amdgcn_s_wait_tensorcnt(0);
    }
    __syncthreads();

    const int kd = (bk % Kk) * Dd + d;
    float Av[Nn];
    #pragma unroll
    for (int n = 0; n < Nn; ++n) Av[n] = -__expf(A_logs[kd * Nn + n]);
    const float dscale = Ds[kd];

    float h[Nn];
    size_t hoff = (((size_t)bk * NC + c) * Dd + d) * Nn;
    #pragma unroll
    for (int n = 0; n < Nn; ++n) h[n] = hin[hoff + n];

    size_t base = ((size_t)bk * Ll + c * CL) * Dd + d;
    for (int t = 0; t < CL; ++t) {
        float dlt = delta_t[base + (size_t)t * Dd];
        float u   = xs_t[base + (size_t)t * Dd];
        float du  = dlt * u;
        const float* bt = lbc + t * N2;
        const float* ct = bt + Nn;
        float y = 0.f;
        #pragma unroll
        for (int n = 0; n < Nn; ++n) {
            float dA = __expf(dlt * Av[n]);
            h[n] = h[n] * dA + du * bt[n];
            y += h[n] * ct[n];
        }
        ys_t[base + (size_t)t * Dd] = y + u * dscale;  // coalesced store
    }
}

// ---------------------------------------------------------------------------
// Kernel 5: cross-merge (8-way inverse-permuted gather-sum) + LayerNorm over D.
//   One block per (b, l); 192 threads = one d each. Output (B,H,W,D).
// ---------------------------------------------------------------------------
__global__ __launch_bounds__(192) void vssm_merge_ln(
    const float* __restrict__ ys_t, const float* __restrict__ ln_w,
    const float* __restrict__ ln_b, float* __restrict__ out)
{
    __shared__ float red[12];
    const int blk = blockIdx.x;            // over B*L
    const int b = blk / Ll, l = blk % Ll;
    const int d = threadIdx.x;

    float acc = 0.f;
    #pragma unroll
    for (int k = 0; k < Kk; ++k) {
        int p = pinv_k(k, l);
        acc += ys_t[(((size_t)(b * Kk + k)) * Ll + p) * Dd + d];
    }
    // block LayerNorm over 192 channels (6 wave32 partials)
    float s = acc, s2 = acc * acc;
    #pragma unroll
    for (int off = 16; off > 0; off >>= 1) {
        s  += __shfl_down(s,  off, 32);
        s2 += __shfl_down(s2, off, 32);
    }
    int wave = d >> 5, lane = d & 31;
    if (lane == 0) { red[wave] = s; red[6 + wave] = s2; }
    __syncthreads();
    if (d == 0) {
        float ts = 0.f, ts2 = 0.f;
        for (int w = 0; w < 6; ++w) { ts += red[w]; ts2 += red[6 + w]; }
        red[0] = ts; red[6] = ts2;
    }
    __syncthreads();
    float mean = red[0] * (1.f / Dd);
    float var  = red[6] * (1.f / Dd) - mean * mean;
    float inv  = rsqrtf(var + 1e-5f);
    out[(size_t)blk * Dd + d] = (acc - mean) * inv * ln_w[d] + ln_b[d];
}

// ---------------------------------------------------------------------------
extern "C" void kernel_launch(void* const* d_in, const int* in_sizes, int n_in,
                              void* d_out, int out_size, void* d_ws, size_t ws_size,
                              hipStream_t stream) {
    const float* x      = (const float*)d_in[0];  // (B,D,H,W)
    const float* xw     = (const float*)d_in[1];  // (K,38,D)
    const float* dtw    = (const float*)d_in[2];  // (K,D,R)
    const float* dtb    = (const float*)d_in[3];  // (K,D)
    const float* A_logs = (const float*)d_in[4];  // (K*D,N)
    const float* Ds     = (const float*)d_in[5];  // (K*D)
    const float* ln_w   = (const float*)d_in[6];
    const float* ln_b   = (const float*)d_in[7];
    float* out = (float*)d_out;

    char* ws = (char*)d_ws;
    size_t o = 0;
    float* xs_t    = (float*)(ws + o); o += (size_t)BK * Ll * Dd * sizeof(float);
    float* delta_t = (float*)(ws + o); o += (size_t)BK * Ll * Dd * sizeof(float);
    float* bc_t    = (float*)(ws + o); o += (size_t)BK * Ll * N2 * sizeof(float);
    float* chunkA  = (float*)(ws + o); o += (size_t)BK * NC * Dd * Nn * sizeof(float);
    float* chunkB  = (float*)(ws + o); o += (size_t)BK * NC * Dd * Nn * sizeof(float);
    float* hin     = (float*)(ws + o); o += (size_t)BK * NC * Dd * Nn * sizeof(float);
    float* ys_t    = (float*)(ws + o); o += (size_t)BK * Ll * Dd * sizeof(float);

    vssm_proj_kernel<<<BK * (Ll / LT), 128, 0, stream>>>(
        x, xw, dtw, dtb, xs_t, delta_t, bc_t);

    vssm_scan_phase1<<<BK * NC, 192, 0, stream>>>(
        xs_t, delta_t, bc_t, A_logs, chunkA, chunkB);

    vssm_scan_phase2<<<(BK * Dd * Nn + 255) / 256, 256, 0, stream>>>(
        chunkA, chunkB, hin);

    vssm_scan_phase3<<<BK * NC, 192, 0, stream>>>(
        xs_t, delta_t, bc_t, A_logs, Ds, hin, ys_t);

    vssm_merge_ln<<<Bb * Ll, 192, 0, stream>>>(ys_t, ln_w, ln_b, out);
}
